// Seq2Seq_11149735100709
// MI455X (gfx1250) — compile-verified, matrix operand order checked by reading, not run on hardware
//
#include <hip/hip_runtime.h>

// ---------------------------------------------------------------------------
// Seq2Seq GRU (encoder/decoder) for MI455X gfx1250, wave32 + WMMA bf16.
// Dominant cost: 992x32000x512 output GEMM (32.5 GFLOP) -> bf16 WMMA,
// Wout (33MB bf16) is L2-resident; logits write (131MB) is the HBM floor.
// GEMM K-loops are software-pipelined with 2x unroll + ping-pong fragment
// buffers (no register rotation copies): loads for K-block i+1 are in
// flight while K-block i's v_wmma issue.
// Recurrence runs in one persistent kernel with an atomic grid barrier.
// ---------------------------------------------------------------------------

typedef __attribute__((ext_vector_type(16))) __bf16 v16bf;
typedef __attribute__((ext_vector_type(8)))  float  v8f;

struct U4 { unsigned x, y, z, w; };
union FragU { v16bf v; U4 q[2]; };

#define NB 32
#define NS 64
#define NT 32
#define NV 32000
#define NE 256
#define NH 512
#define G3 1536   // 3*H

// ---- helpers --------------------------------------------------------------

__device__ inline unsigned short f32_to_bf16(float f) {
    unsigned u = __float_as_uint(f);
    u += 0x7FFFu + ((u >> 16) & 1u);           // round-to-nearest-even
    return (unsigned short)(u >> 16);
}

__device__ inline float fast_sigm(float x) { return 1.0f / (1.0f + __expf(-x)); }
__device__ inline float fast_tanh(float x) {
    float e = __expf(-2.0f * x);
    return (1.0f - e) / (1.0f + e);
}

// bf16 fragment loader for 16x32 A tile (row = M) or 32x16 B tile (row = N),
// source row-major with leading dim = lda (elements), 16B-aligned rows.
// 16-bit operand layout (ISA 7.12.2): halves 0..7 -> K = kb+8g.., halves
// 8..15 -> K = kb+16+8g.. ; two contiguous 16B groups per lane.
__device__ inline v16bf load_frag(const unsigned short* base, int row0, int lda,
                                  int kb, int lane) {
    const int g = lane >> 4;
    const unsigned short* p =
        base + (size_t)(row0 + (lane & 15)) * (size_t)lda + kb + g * 8;
    FragU f;
    f.q[0] = *(const U4*)(p);        // K = kb + 8g .. +7
    f.q[1] = *(const U4*)(p + 16);   // K = kb + 16 + 8g .. +7
    return f.v;
}

__device__ inline v8f wmma_bf16(v16bf a, v16bf b, v8f c) {
    return __builtin_amdgcn_wmma_f32_16x16x32_bf16(
        false, a, false, b, (short)0, c, false, false);
}

// ---- small utility kernels ------------------------------------------------

__global__ __launch_bounds__(256) void cvt_bf16_kernel(const float* __restrict__ src,
                                                       unsigned short* __restrict__ dst,
                                                       int n) {
    for (int i = blockIdx.x * blockDim.x + threadIdx.x; i < n;
         i += gridDim.x * blockDim.x)
        dst[i] = f32_to_bf16(src[i]);
}

// x[m, e] = emb[tok(m), e] as bf16; m = step*32 + b; tok = toks[b*tokStride + step]
__global__ __launch_bounds__(256) void embed_gather_kernel(const float* __restrict__ emb,
                                                           const int* __restrict__ toks,
                                                           unsigned short* __restrict__ X,
                                                           int rows, int tokStride) {
    int total = rows * NE;
    for (int i = blockIdx.x * blockDim.x + threadIdx.x; i < total;
         i += gridDim.x * blockDim.x) {
        int m = i >> 8;            // / NE
        int e = i & (NE - 1);
        int b = m & 31;
        int s = m >> 5;
        int tok = toks[b * tokStride + s];
        X[i] = f32_to_bf16(emb[(size_t)tok * NE + e]);
    }
}

__global__ __launch_bounds__(256) void init_state_kernel(unsigned* sync2,
                                                         float* h,
                                                         unsigned short* h_bf) {
    if (threadIdx.x < 2) sync2[threadIdx.x] = 0u;
    for (int i = threadIdx.x; i < NB * NH; i += blockDim.x) {
        h[i] = 0.0f;
        h_bf[i] = 0;
    }
}

__global__ __launch_bounds__(256) void zero_t0_kernel(float* __restrict__ out) {
    int total = NB * NV;
    for (int i = blockIdx.x * blockDim.x + threadIdx.x; i < total;
         i += gridDim.x * blockDim.x) {
        int b = i / NV, v = i - b * NV;
        out[(size_t)b * NT * NV + v] = 0.0f;
    }
}

// ---- generic WMMA GEMM: C[M,N] = A[M,K](bf16) * W[N,K]^T(bf16) + bias -----
// block = 256 threads (8 waves); block tile 32(M) x 256(N).
// Wave tile 2M x 2N (32x32): each wave owns a unique 32-column B slice, so
// every B fragment is loaded exactly once per block; A (tiny, L2-hot) is
// duplicated across waves.
// K pipeline: 2x unrolled ping-pong fragment buffers -> loads for block
// i+1 overlap WMMAs of block i, with no register-rotation copies.
// Requires K % 64 == 0 (K is 256 or 512 here).
// mode 0: C row-major (ld = N).  mode 1: logits scatter out[b, t+1, n].
__global__ __launch_bounds__(256)
void wmma_gemm_bias(const unsigned short* __restrict__ A,
                    const unsigned short* __restrict__ W,
                    const float* __restrict__ bias,
                    float* __restrict__ C,
                    int N, int K, int mode) {
    const int lane = threadIdx.x & 31;
    const int wv   = threadIdx.x >> 5;
    const int m0 = blockIdx.y * 32;
    const int n0 = blockIdx.x * 256 + wv * 32;

    v8f zero = {};
    v8f acc[2][2];
#pragma unroll
    for (int mt = 0; mt < 2; ++mt)
#pragma unroll
        for (int nt = 0; nt < 2; ++nt) acc[mt][nt] = zero;

    v16bf aP[2], bP[2];   // ping buffer
    v16bf aQ[2], bQ[2];   // pong buffer
    aP[0] = load_frag(A, m0,      K, 0, lane);
    aP[1] = load_frag(A, m0 + 16, K, 0, lane);
    bP[0] = load_frag(W, n0,      K, 0, lane);
    bP[1] = load_frag(W, n0 + 16, K, 0, lane);

    int kb = 0;
    for (; kb + 64 < K; kb += 64) {
        aQ[0] = load_frag(A, m0,      K, kb + 32, lane);
        aQ[1] = load_frag(A, m0 + 16, K, kb + 32, lane);
        bQ[0] = load_frag(W, n0,      K, kb + 32, lane);
        bQ[1] = load_frag(W, n0 + 16, K, kb + 32, lane);
#pragma unroll
        for (int mt = 0; mt < 2; ++mt)
#pragma unroll
            for (int nt = 0; nt < 2; ++nt)
                acc[mt][nt] = wmma_bf16(aP[mt], bP[nt], acc[mt][nt]);

        aP[0] = load_frag(A, m0,      K, kb + 64, lane);
        aP[1] = load_frag(A, m0 + 16, K, kb + 64, lane);
        bP[0] = load_frag(W, n0,      K, kb + 64, lane);
        bP[1] = load_frag(W, n0 + 16, K, kb + 64, lane);
#pragma unroll
        for (int mt = 0; mt < 2; ++mt)
#pragma unroll
            for (int nt = 0; nt < 2; ++nt)
                acc[mt][nt] = wmma_bf16(aQ[mt], bQ[nt], acc[mt][nt]);
    }
    // tail: kb == K - 64; process kb (in P) and kb+32 (in Q)
    aQ[0] = load_frag(A, m0,      K, kb + 32, lane);
    aQ[1] = load_frag(A, m0 + 16, K, kb + 32, lane);
    bQ[0] = load_frag(W, n0,      K, kb + 32, lane);
    bQ[1] = load_frag(W, n0 + 16, K, kb + 32, lane);
#pragma unroll
    for (int mt = 0; mt < 2; ++mt)
#pragma unroll
        for (int nt = 0; nt < 2; ++nt)
            acc[mt][nt] = wmma_bf16(aP[mt], bP[nt], acc[mt][nt]);
#pragma unroll
    for (int mt = 0; mt < 2; ++mt)
#pragma unroll
        for (int nt = 0; nt < 2; ++nt)
            acc[mt][nt] = wmma_bf16(aQ[mt], bQ[nt], acc[mt][nt]);

    const int nc = lane & 15;
    const int mo = (lane >> 4) << 3;          // C layout: M = vgpr + 8*(lane/16)
#pragma unroll
    for (int nt = 0; nt < 2; ++nt) {
        int n = n0 + nt * 16 + nc;
        float bv = bias[n];
#pragma unroll
        for (int mt = 0; mt < 2; ++mt) {
#pragma unroll
            for (int v = 0; v < 8; ++v) {
                int m = m0 + mt * 16 + mo + v;
                float val = acc[mt][nt][v] + bv;
                if (mode == 0) {
                    C[(size_t)m * (size_t)N + n] = val;
                } else {
                    // m = t*32 + b  ->  out[b, t+1, n]
                    int t = m >> 5, b = m & 31;
                    C[((size_t)b * NT + (t + 1)) * (size_t)NV + n] = val;
                }
            }
        }
    }
}

// ---- persistent recurrent kernel ------------------------------------------

__device__ inline void grid_sync(unsigned* cnt, unsigned* gen, int nwg,
                                 unsigned* sGen) {
    __threadfence();
    __syncthreads();
    if (threadIdx.x == 0) {
        unsigned g = *sGen;
        unsigned prev = __hip_atomic_fetch_add(cnt, 1u, __ATOMIC_ACQ_REL,
                                               __HIP_MEMORY_SCOPE_AGENT);
        if (prev == (unsigned)nwg - 1u) {
            __hip_atomic_store(cnt, 0u, __ATOMIC_RELAXED,
                               __HIP_MEMORY_SCOPE_AGENT);
            __hip_atomic_fetch_add(gen, 1u, __ATOMIC_RELEASE,
                                   __HIP_MEMORY_SCOPE_AGENT);
        } else {
            while (__hip_atomic_load(gen, __ATOMIC_ACQUIRE,
                                     __HIP_MEMORY_SCOPE_AGENT) == g)
                __builtin_amdgcn_s_sleep(1);
        }
        *sGen = g + 1u;
    }
    __syncthreads();
}

// gh[M=32, Ntiles*16] = h_bf(32xK) @ Wbf(N,K)^T ; tiles distributed over
// waves; K pipeline 2x unrolled ping-pong (K % 64 == 0; K = 512 here).
__device__ inline void gemm_tiles(const unsigned short* Abf,
                                  const unsigned short* Wbf,
                                  float* Cout, int Mtiles, int Ntiles,
                                  int K, int ldc, int wv, int nWaves) {
    const int lane = threadIdx.x & 31;
    const int total = Mtiles * Ntiles;
    for (int tIdx = wv; tIdx < total; tIdx += nWaves) {
        int tm = tIdx % Mtiles, tn = tIdx / Mtiles;
        int m0 = tm * 16, n0 = tn * 16;
        v8f acc = {};
        v16bf aP = load_frag(Abf, m0, K, 0, lane);
        v16bf bP = load_frag(Wbf, n0, K, 0, lane);
        int kb = 0;
        for (; kb + 64 < K; kb += 64) {
            v16bf aQ = load_frag(Abf, m0, K, kb + 32, lane);
            v16bf bQ = load_frag(Wbf, n0, K, kb + 32, lane);
            acc = wmma_bf16(aP, bP, acc);
            aP = load_frag(Abf, m0, K, kb + 64, lane);
            bP = load_frag(Wbf, n0, K, kb + 64, lane);
            acc = wmma_bf16(aQ, bQ, acc);
        }
        v16bf aQ = load_frag(Abf, m0, K, kb + 32, lane);
        v16bf bQ = load_frag(Wbf, n0, K, kb + 32, lane);
        acc = wmma_bf16(aP, bP, acc);
        acc = wmma_bf16(aQ, bQ, acc);

        int nc = n0 + (lane & 15);
        int mb = m0 + ((lane >> 4) << 3);
#pragma unroll
        for (int v = 0; v < 8; ++v)
            Cout[(size_t)(mb + v) * (size_t)ldc + nc] = acc[v];
    }
}

__global__ __launch_bounds__(256)
void gru_persistent(const float* __restrict__ gi_enc,   // (S*32, 1536) incl bih
                    const float* __restrict__ gi_dec,   // (31*32, 1536) incl bih
                    const unsigned short* __restrict__ Whh_enc,
                    const unsigned short* __restrict__ Whh_dec,
                    const float* __restrict__ enc_bhh,
                    const float* __restrict__ dec_bhh,
                    const unsigned short* __restrict__ Wt_bf,
                    const float* __restrict__ enc_bt,
                    const int* __restrict__ src_len,
                    float* __restrict__ gh,             // (32,1536) scratch
                    float* __restrict__ h,              // (32,512)
                    unsigned short* __restrict__ h_bf,  // (32,512)
                    unsigned short* __restrict__ hs_bf, // (31*32,512)
                    unsigned* sync_cnt, unsigned* sync_gen, int nwg) {
    __shared__ unsigned sGen;
    if (threadIdx.x == 0)
        sGen = __hip_atomic_load(sync_gen, __ATOMIC_ACQUIRE,
                                 __HIP_MEMORY_SCOPE_AGENT);
    __syncthreads();

    const int nWaves = nwg * (blockDim.x >> 5);
    const int wv = (blockIdx.x * blockDim.x + threadIdx.x) >> 5;
    const int gtid = blockIdx.x * blockDim.x + threadIdx.x;
    const int nThreads = nwg * blockDim.x;

    // ---------------- encoder scan: 64 masked GRU steps ----------------
    for (int s = 0; s < NS; ++s) {
        gemm_tiles(h_bf, Whh_enc, gh, 2, G3 / 16, NH, G3, wv, nWaves);
        grid_sync(sync_cnt, sync_gen, nwg, &sGen);

        for (int idx = gtid; idx < NB * NH; idx += nThreads) {
            int b = idx >> 9, j = idx & (NH - 1);
            const float* gi = gi_enc + (size_t)(s * NB + b) * G3;
            const float* gr = gh + (size_t)b * G3;
            float r = fast_sigm(gi[j]            + gr[j]            + enc_bhh[j]);
            float z = fast_sigm(gi[NH + j]       + gr[NH + j]       + enc_bhh[NH + j]);
            float hn = gr[2 * NH + j] + enc_bhh[2 * NH + j];
            float n = fast_tanh(gi[2 * NH + j] + r * hn);
            float hold = h[idx];
            float hnew = (1.0f - z) * n + z * hold;
            if (s >= src_len[b]) hnew = hold;   // mask: keep old past src_len
            h[idx] = hnew;
            h_bf[idx] = f32_to_bf16(hnew);
        }
        grid_sync(sync_cnt, sync_gen, nwg, &sGen);
    }

    // ---------------- hidden0 = tanh(h_enc @ Wt^T + bt) ----------------
    gemm_tiles(h_bf, Wt_bf, gh, 2, NH / 16, NH, NH, wv, nWaves);
    grid_sync(sync_cnt, sync_gen, nwg, &sGen);
    for (int idx = gtid; idx < NB * NH; idx += nThreads) {
        int j = idx & (NH - 1);
        float v = fast_tanh(gh[idx] + enc_bt[j]);
        h[idx] = v;
        h_bf[idx] = f32_to_bf16(v);
    }
    grid_sync(sync_cnt, sync_gen, nwg, &sGen);

    // ---------------- decoder scan: 31 GRU steps, record hs ------------
    for (int t = 0; t < NT - 1; ++t) {
        gemm_tiles(h_bf, Whh_dec, gh, 2, G3 / 16, NH, G3, wv, nWaves);
        grid_sync(sync_cnt, sync_gen, nwg, &sGen);

        for (int idx = gtid; idx < NB * NH; idx += nThreads) {
            int b = idx >> 9, j = idx & (NH - 1);
            const float* gi = gi_dec + (size_t)(t * NB + b) * G3;
            const float* gr = gh + (size_t)b * G3;
            float r = fast_sigm(gi[j]      + gr[j]      + dec_bhh[j]);
            float z = fast_sigm(gi[NH + j] + gr[NH + j] + dec_bhh[NH + j]);
            float hn = gr[2 * NH + j] + dec_bhh[2 * NH + j];
            float n = fast_tanh(gi[2 * NH + j] + r * hn);
            float hnew = (1.0f - z) * n + z * h[idx];
            h[idx] = hnew;
            unsigned short hb = f32_to_bf16(hnew);
            h_bf[idx] = hb;
            hs_bf[(size_t)t * (NB * NH) + idx] = hb;  // row = t*32 + b
        }
        grid_sync(sync_cnt, sync_gen, nwg, &sGen);
    }
}

// ---------------------------------------------------------------------------

extern "C" void kernel_launch(void* const* d_in, const int* in_sizes, int n_in,
                              void* d_out, int out_size, void* d_ws, size_t ws_size,
                              hipStream_t stream) {
    (void)in_sizes; (void)n_in; (void)out_size; (void)ws_size;

    const int*   src      = (const int*)  d_in[0];
    const int*   src_len  = (const int*)  d_in[1];
    const int*   trg      = (const int*)  d_in[2];
    const float* enc_emb  = (const float*)d_in[3];
    const float* enc_Wih  = (const float*)d_in[4];
    const float* enc_Whh  = (const float*)d_in[5];
    const float* enc_bih  = (const float*)d_in[6];
    const float* enc_bhh  = (const float*)d_in[7];
    const float* enc_Wt   = (const float*)d_in[8];
    const float* enc_bt   = (const float*)d_in[9];
    const float* dec_emb  = (const float*)d_in[10];
    const float* dec_Wih  = (const float*)d_in[11];
    const float* dec_Whh  = (const float*)d_in[12];
    const float* dec_bih  = (const float*)d_in[13];
    const float* dec_bhh  = (const float*)d_in[14];
    const float* dec_Wout = (const float*)d_in[15];
    const float* dec_bout = (const float*)d_in[16];
    float* out = (float*)d_out;

    // ---- workspace carve-out (256B aligned) ----
    char* w = (char*)d_ws;
    size_t off = 0;
    auto take = [&](size_t bytes) -> char* {
        char* r = w + off;
        off += (bytes + 255) & ~(size_t)255;
        return r;
    };
    unsigned*       sync2      = (unsigned*)      take(256);
    float*          h          = (float*)         take((size_t)NB * NH * 4);
    unsigned short* h_bf       = (unsigned short*)take((size_t)NB * NH * 2);
    float*          gh         = (float*)         take((size_t)NB * G3 * 4);
    unsigned short* x_enc      = (unsigned short*)take((size_t)NS * NB * NE * 2);
    unsigned short* x_dec      = (unsigned short*)take((size_t)(NT - 1) * NB * NE * 2);
    float*          gi_enc     = (float*)         take((size_t)NS * NB * G3 * 4);
    float*          gi_dec     = (float*)         take((size_t)(NT - 1) * NB * G3 * 4);
    unsigned short* Wih_enc_bf = (unsigned short*)take((size_t)G3 * NE * 2);
    unsigned short* Whh_enc_bf = (unsigned short*)take((size_t)G3 * NH * 2);
    unsigned short* Wih_dec_bf = (unsigned short*)take((size_t)G3 * NE * 2);
    unsigned short* Whh_dec_bf = (unsigned short*)take((size_t)G3 * NH * 2);
    unsigned short* Wt_bf      = (unsigned short*)take((size_t)NH * NH * 2);
    unsigned short* Wout_bf    = (unsigned short*)take((size_t)NV * NH * 2);
    unsigned short* hs_bf      = (unsigned short*)take((size_t)(NT - 1) * NB * NH * 2);

    const int TPB = 256;
    auto blocksFor = [](int n) {
        int b = (n + 255) / 256;
        return b > 2048 ? 2048 : b;
    };

    // 0) reset sync + zero h0
    init_state_kernel<<<1, TPB, 0, stream>>>(sync2, h, h_bf);

    // 1) weight conversions fp32 -> bf16
    cvt_bf16_kernel<<<blocksFor(G3 * NE), TPB, 0, stream>>>(enc_Wih, Wih_enc_bf, G3 * NE);
    cvt_bf16_kernel<<<blocksFor(G3 * NH), TPB, 0, stream>>>(enc_Whh, Whh_enc_bf, G3 * NH);
    cvt_bf16_kernel<<<blocksFor(G3 * NE), TPB, 0, stream>>>(dec_Wih, Wih_dec_bf, G3 * NE);
    cvt_bf16_kernel<<<blocksFor(G3 * NH), TPB, 0, stream>>>(dec_Whh, Whh_dec_bf, G3 * NH);
    cvt_bf16_kernel<<<blocksFor(NH * NH), TPB, 0, stream>>>(enc_Wt, Wt_bf, NH * NH);
    cvt_bf16_kernel<<<blocksFor(NV * NH), TPB, 0, stream>>>(dec_Wout, Wout_bf, NV * NH);

    // 2) embedding gathers (row m = step*32 + b)
    embed_gather_kernel<<<blocksFor(NS * NB * NE), TPB, 0, stream>>>(
        enc_emb, src, x_enc, NS * NB, NS);
    embed_gather_kernel<<<blocksFor((NT - 1) * NB * NE), TPB, 0, stream>>>(
        dec_emb, trg, x_dec, (NT - 1) * NB, NT);

    // 3) input-side GEMMs: gi = x @ Wih^T + bih  (hoisted out of the scans)
    {
        dim3 g(G3 / 256, (NS * NB) / 32);
        wmma_gemm_bias<<<g, TPB, 0, stream>>>(x_enc, Wih_enc_bf, enc_bih,
                                              gi_enc, G3, NE, 0);
    }
    {
        dim3 g(G3 / 256, ((NT - 1) * NB) / 32);
        wmma_gemm_bias<<<g, TPB, 0, stream>>>(x_dec, Wih_dec_bf, dec_bih,
                                              gi_dec, G3, NE, 0);
    }

    // 4) persistent recurrent scan (encoder + bridge + decoder)
    const int NWG = 16;
    gru_persistent<<<NWG, TPB, 0, stream>>>(
        gi_enc, gi_dec, Whh_enc_bf, Whh_dec_bf, enc_bhh, dec_bhh,
        Wt_bf, enc_bt, src_len, gh, h, h_bf, hs_bf,
        sync2 + 0, sync2 + 1, NWG);

    // 5) outputs: zero t=0 slice, then logits GEMM with fused scatter+bias
    zero_t0_kernel<<<blocksFor(NB * NV), TPB, 0, stream>>>(out);
    {
        dim3 g(NV / 256, ((NT - 1) * NB) / 32);  // 125 x 31 blocks
        wmma_gemm_bias<<<g, TPB, 0, stream>>>(hs_bf, Wout_bf, dec_bout,
                                              out, NV, NH, 1);
    }
}
